// EFIN_43868795961894
// MI455X (gfx1250) — compile-verified, matrix operand order checked by reading, not run on hardware
//
#include <hip/hip_runtime.h>
#include <hip/hip_bf16.h>

// ---------------------------------------------------------------------------
// EFIN uplift net on MI455X (gfx1250): fp16 WMMA GEMMs (64x32 register-blocked)
// + fused LDS attention + global_prefetch_b8 on streaming A operands.
// ---------------------------------------------------------------------------

typedef __attribute__((ext_vector_type(16))) _Float16 v16h;
typedef __attribute__((ext_vector_type(8)))  _Float16 v8h;
typedef __attribute__((ext_vector_type(8)))  float    v8f;

#define EB   2048
#define EF   100
#define EHU  128
#define EHC  512
#define EBF  (EB * EF)          // 204800
#define EBF_PAD (EBF + 32)      // slack rows so padded attention row-tiles can read

// -------- fragment loader: 16-bit A-matrix layout (ISA 7.12.2) --------------
// lane l: row = l&15, half = l>>4; element e<8 -> k = half*8+e,
// e>=8 -> k = 16 + half*8 + (e-8).  Two contiguous 16B loads per lane.
__device__ __forceinline__ v16h frag_ld(const _Float16* __restrict__ row,
                                        int half, int k0) {
  v8h lo = *(const v8h*)(row + k0 + half * 8);
  v8h hi = *(const v8h*)(row + k0 + 16 + half * 8);
  v16h r;
#pragma unroll
  for (int i = 0; i < 8; ++i) { r[i] = lo[i]; r[i + 8] = hi[i]; }
  return r;
}

__device__ __forceinline__ float sigmoidf(float x) {
  return 1.0f / (1.0f + __expf(-x));
}
__device__ __forceinline__ float eluf(float x) {
  return x > 0.0f ? x : (__expf(x) - 1.0f);
}

// -------- register-blocked WMMA GEMM ----------------------------------------
// C[M,N] = A[M,K](row) * W[N,K](row)^T + bias.  One wave computes a 64x32
// block (4 M-tiles x 2 N-tiles): per K-step 6 fragment loads feed 8 WMMAs
// (~21 flop/byte vs 8 unblocked).  Requires 64|M, 32|N, 32|K.
// act: 0 = none, 1 = ELU (stored activation consumed by the next layer).
__global__ void gemm_rrr_f16_blk(const _Float16* __restrict__ A, int lda,
                                 const _Float16* __restrict__ W, int ldw,
                                 const float* __restrict__ bias,
                                 _Float16* __restrict__ C, int ldc,
                                 int M, int N, int K, int act) {
  int wave = (int)((blockIdx.x * blockDim.x + threadIdx.x) >> 5);
  int lane = threadIdx.x & 31;
  int tn = N >> 5;                       // 32-wide N blocks
  int tiles = (M >> 6) * tn;             // 64-tall M blocks
  if (wave >= tiles) return;             // wave-uniform
  int bm = wave / tn, bn = wave % tn;
  int half = lane >> 4, l16 = lane & 15;

  const _Float16* arow[4];
#pragma unroll
  for (int i = 0; i < 4; ++i)
    arow[i] = A + (size_t)(bm * 64 + i * 16 + l16) * lda;
  const _Float16* wrow[2];
#pragma unroll
  for (int j = 0; j < 2; ++j)
    wrow[j] = W + (size_t)(bn * 32 + j * 16 + l16) * ldw;

  v8f acc[4][2];
#pragma unroll
  for (int i = 0; i < 4; ++i)
#pragma unroll
    for (int j = 0; j < 2; ++j)
#pragma unroll
      for (int v = 0; v < 8; ++v) acc[i][j][v] = 0.0f;

  for (int k0 = 0; k0 < K; k0 += 32) {
    // prefetch the A streams a few K-steps ahead (lowers to global_prefetch_b8)
#pragma unroll
    for (int i = 0; i < 4; ++i)
      __builtin_prefetch(arow[i] + k0 + 128, 0, 1);
    v16h a[4];
#pragma unroll
    for (int i = 0; i < 4; ++i) a[i] = frag_ld(arow[i], half, k0);
    v16h b[2];
#pragma unroll
    for (int j = 0; j < 2; ++j) b[j] = frag_ld(wrow[j], half, k0);
#pragma unroll
    for (int i = 0; i < 4; ++i)
#pragma unroll
      for (int j = 0; j < 2; ++j)
        acc[i][j] = __builtin_amdgcn_wmma_f32_16x16x32_f16(
            false, a[i], false, b[j], (short)0, acc[i][j], false, false);
  }

#pragma unroll
  for (int j = 0; j < 2; ++j) {
    int col = bn * 32 + j * 16 + l16;
    float bv = bias ? bias[col] : 0.0f;
#pragma unroll
    for (int i = 0; i < 4; ++i) {
#pragma unroll
      for (int v = 0; v < 8; ++v) {
        int row = bm * 64 + i * 16 + v + 8 * half;
        float x = acc[i][j][v] + bv;
        if (act) x = eluf(x);
        C[(size_t)row * ldc + col] = (_Float16)x;
      }
    }
  }
}

// -------- WMMA GEMM with K-major B: C[M,N] = A[M,K] * B[K,N] ----------------
// (only used for the tiny xt = wf @ E_pad GEMM: 2048x128x128)
__global__ void gemm_bk_f16(const _Float16* __restrict__ A, int lda,
                            const _Float16* __restrict__ Bm, int ldb,
                            _Float16* __restrict__ C, int ldc,
                            int M, int N, int K) {
  int wave = (int)((blockIdx.x * blockDim.x + threadIdx.x) >> 5);
  int lane = threadIdx.x & 31;
  int tn = N >> 4;
  int tiles = (M >> 4) * tn;
  if (wave >= tiles) return;
  int tmi = wave / tn, tni = wave % tn;
  int half = lane >> 4, l16 = lane & 15;
  const _Float16* arow = A + (size_t)(tmi * 16 + l16) * lda;
  int ncol = tni * 16 + l16;
  v8f acc;
#pragma unroll
  for (int i = 0; i < 8; ++i) acc[i] = 0.0f;
  for (int k0 = 0; k0 < K; k0 += 32) {
    v16h a = frag_ld(arow, half, k0);
    v16h b;
#pragma unroll
    for (int e = 0; e < 16; ++e) {
      int k = k0 + ((e < 8) ? (half * 8 + e) : (16 + half * 8 + (e - 8)));
      b[e] = Bm[(size_t)k * ldb + ncol];
    }
    acc = __builtin_amdgcn_wmma_f32_16x16x32_f16(false, a, false, b,
                                                 (short)0, acc, false, false);
  }
#pragma unroll
  for (int v = 0; v < 8; ++v) {
    int row = tmi * 16 + v + 8 * half;
    C[(size_t)row * ldc + ncol] = (_Float16)acc[v];
  }
}

// -------- small helpers -----------------------------------------------------
__global__ void f32_to_f16_kern(const float* __restrict__ s,
                                _Float16* __restrict__ d, int n) {
  int i = blockIdx.x * blockDim.x + threadIdx.x;
  if (i < n) d[i] = (_Float16)s[i];
}

// rnorm[b,h] = 1/sqrt(sum_f (feat[b,f]*E[f,h])^2)
__global__ void norm_kern(const float* __restrict__ feat,
                          const float* __restrict__ E,
                          float* __restrict__ rnorm) {
  int i = blockIdx.x * blockDim.x + threadIdx.x;
  if (i >= EB * EHU) return;
  int b = i >> 7, h = i & 127;
  float s = 0.0f;
  for (int f = 0; f < EF; ++f) {
    float t = feat[b * EF + f] * E[f * EHU + h];
    s += t * t;
  }
  rnorm[i] = rsqrtf(s);
}

// nx[(b*F+f)*HU+h] = feat*E*rnorm  (fp16)
__global__ void nx_kern(const float* __restrict__ feat,
                        const float* __restrict__ E,
                        const float* __restrict__ rnorm,
                        _Float16* __restrict__ nx) {
  long long i = (long long)blockIdx.x * blockDim.x + threadIdx.x;
  if (i >= (long long)EBF * EHU) return;
  int h = (int)(i & 127);
  long long bf = i >> 7;
  int f = (int)(bf % EF);
  int b = (int)(bf / EF);
  nx[i] = (_Float16)(feat[b * EF + f] * E[f * EHU + h] * rnorm[b * EHU + h]);
}

// -------- fused attention: one block per batch, 7 waves (224 threads) -------
// scores = (Q K^T)/sqrt(HU); attn = softmax(sigmoid(scores)); xx = attn V
__global__ __launch_bounds__(224)
void attn_kern(const _Float16* __restrict__ Q,
               const _Float16* __restrict__ Km,
               const _Float16* __restrict__ V,
               _Float16* __restrict__ XX) {
  __shared__ __align__(16) _Float16 sAttn[112 * 128];  // rows 0..111, K pad 128
  __shared__ __align__(16) _Float16 sVt[128 * 128];    // Vt[h][f], f pad zeroed
  int b = blockIdx.x;
  int tid = threadIdx.x;

  // transpose V into LDS, zero the f>=100 pad
  for (int idx = tid; idx < 128 * 128; idx += 224) {
    int f = idx >> 7, h = idx & 127;
    _Float16 v = (_Float16)0.0f;
    if (f < EF) v = V[((size_t)b * EF + f) * EHU + h];
    sVt[h * 128 + f] = v;
  }

  int wave = tid >> 5;              // row tile 0..6
  int lane = tid & 31;
  int half = lane >> 4, l16 = lane & 15;

  // ---- scores: 7 column tiles per wave, K = HU = 128 ----
  const _Float16* qrow = Q + ((size_t)b * EF + wave * 16 + l16) * EHU;
  v8f acc[7];
#pragma unroll
  for (int t = 0; t < 7; ++t)
#pragma unroll
    for (int i = 0; i < 8; ++i) acc[t][i] = 0.0f;

  for (int k0 = 0; k0 < EHU; k0 += 32) {
    v16h a = frag_ld(qrow, half, k0);
#pragma unroll
    for (int t = 0; t < 7; ++t) {
      const _Float16* krow = Km + ((size_t)b * EF + t * 16 + l16) * EHU;
      v16h bb = frag_ld(krow, half, k0);
      acc[t] = __builtin_amdgcn_wmma_f32_16x16x32_f16(false, a, false, bb,
                                                      (short)0, acc[t],
                                                      false, false);
    }
  }

  // ---- sigmoid + softmax over columns (row lives in 16 lanes of one half) --
  const float scale = 0.08838834764831845f;   // 1/sqrt(128)
  float e[7][8];
  float rsum[8];
#pragma unroll
  for (int v = 0; v < 8; ++v) rsum[v] = 0.0f;
#pragma unroll
  for (int t = 0; t < 7; ++t) {
    int col = t * 16 + l16;
#pragma unroll
    for (int v = 0; v < 8; ++v) {
      float sg = sigmoidf(acc[t][v] * scale);
      float ev = __expf(sg);                  // in (1, e): safe, no max needed
      if (col >= EF) ev = 0.0f;
      e[t][v] = ev;
      rsum[v] += ev;
    }
  }
#pragma unroll
  for (int v = 0; v < 8; ++v) {
    rsum[v] += __shfl_xor(rsum[v], 1);
    rsum[v] += __shfl_xor(rsum[v], 2);
    rsum[v] += __shfl_xor(rsum[v], 4);
    rsum[v] += __shfl_xor(rsum[v], 8);
    rsum[v] = 1.0f / rsum[v];
  }

  // ---- stage attn rows into LDS (K padded to 128 with zeros) ----
#pragma unroll
  for (int t = 0; t < 7; ++t) {
    int col = t * 16 + l16;
#pragma unroll
    for (int v = 0; v < 8; ++v) {
      int row = wave * 16 + v + 8 * half;
      sAttn[row * 128 + col] = (_Float16)(e[t][v] * rsum[v]);
    }
  }
  {
    int col = 112 + l16;
#pragma unroll
    for (int v = 0; v < 8; ++v) {
      int row = wave * 16 + v + 8 * half;
      sAttn[row * 128 + col] = (_Float16)0.0f;
    }
  }
  __syncthreads();

  // ---- xx = attn @ V  (A from sAttn, B rows from sVt) ----
  const _Float16* arow = sAttn + (wave * 16 + l16) * 128;
#pragma unroll
  for (int t = 0; t < 8; ++t) {
    const _Float16* wrow = sVt + (t * 16 + l16) * 128;
    v8f o;
#pragma unroll
    for (int i = 0; i < 8; ++i) o[i] = 0.0f;
    for (int k0 = 0; k0 < 128; k0 += 32) {
      v16h a = frag_ld(arow, half, k0);
      v16h bb = frag_ld(wrow, half, k0);
      o = __builtin_amdgcn_wmma_f32_16x16x32_f16(false, a, false, bb,
                                                 (short)0, o, false, false);
    }
    int col = t * 16 + l16;
#pragma unroll
    for (int v = 0; v < 8; ++v) {
      int row = wave * 16 + v + 8 * half;
      if (row < EF)
        XX[(size_t)b * (EF * EHU) + (size_t)row * EHU + col] = (_Float16)o[v];
    }
  }
}

// -------- att path (algebraically reduced; x_rep never materialized) --------
// att_t[h] = sigmoid(sum_k (Tw[k]+Tb[k]) * A1[h,k])   (constant over batch)
__global__ void att_t_kern(const float* __restrict__ Tw,
                           const float* __restrict__ Tb,
                           const float* __restrict__ A1,
                           float* __restrict__ attT) {
  int h = threadIdx.x;
  float s = 0.0f;
  for (int k = 0; k < EHU; ++k) s += (Tw[k] + Tb[k]) * A1[h * EHU + k];
  attT[h] = sigmoidf(s);
}

// EA2[f,h] = sum_k E[f,k] * A2w[h,k]
__global__ void ea2_kern(const float* __restrict__ E,
                         const float* __restrict__ A2w,
                         float* __restrict__ EA2) {
  int i = blockIdx.x * blockDim.x + threadIdx.x;
  if (i >= EF * EHU) return;
  int f = i >> 7, h = i & 127;
  float s = 0.0f;
  for (int k = 0; k < EHU; ++k) s += E[f * EHU + k] * A2w[h * EHU + k];
  EA2[i] = s;
}

// logits[b,f] = sum_h relu(attT[h] + sigmoid(feat*EA2[f,h]+A2b[h])) * A3[h]
__global__ void logits_kern(const float* __restrict__ feat,
                            const float* __restrict__ EA2,
                            const float* __restrict__ A2b,
                            const float* __restrict__ attT,
                            const float* __restrict__ A3,
                            float* __restrict__ logits) {
  int i = blockIdx.x * blockDim.x + threadIdx.x;
  if (i >= EBF) return;
  int f = i % EF;
  float x = feat[i];
  float s = 0.0f;
  for (int h = 0; h < EHU; ++h) {
    float r = attT[h] + sigmoidf(x * EA2[f * EHU + h] + A2b[h]);
    if (r > 0.0f) s += r * A3[h];
  }
  logits[i] = s;
}

// w = softmax_f(logits); wf[b,f] = w * feat[b,f]  (fp16, padded to 128)
__global__ void wsoft_kern(const float* __restrict__ logits,
                           const float* __restrict__ feat,
                           _Float16* __restrict__ wf) {
  int b = blockIdx.x * blockDim.x + threadIdx.x;
  if (b >= EB) return;
  float mx = -1e30f;
  for (int f = 0; f < EF; ++f) mx = fmaxf(mx, logits[b * EF + f]);
  float s = 0.0f;
  for (int f = 0; f < EF; ++f) s += __expf(logits[b * EF + f] - mx);
  float inv = 1.0f / s;
  for (int f = 0; f < EF; ++f)
    wf[b * 128 + f] =
        (_Float16)(__expf(logits[b * EF + f] - mx) * inv * feat[b * EF + f]);
  for (int f = EF; f < 128; ++f) wf[b * 128 + f] = (_Float16)0.0f;
}

// E (100x128 f32) -> fp16, rows 100..127 zeroed (K-major B for xt GEMM)
__global__ void epad_kern(const float* __restrict__ E, _Float16* __restrict__ Ep) {
  int i = blockIdx.x * blockDim.x + threadIdx.x;
  if (i >= 128 * 128) return;
  int f = i >> 7, h = i & 127;
  Ep[i] = (_Float16)((f < EF) ? E[f * EHU + h] : 0.0f);
}

// -------- heads: 6 outputs ---------------------------------------------------
__global__ void heads_kern(const _Float16* __restrict__ c_last,
                           const _Float16* __restrict__ u_last,
                           const float* __restrict__ clw, const float* __restrict__ clb,
                           const float* __restrict__ ctw, const float* __restrict__ ctb,
                           const float* __restrict__ tlw, const float* __restrict__ tlb,
                           const float* __restrict__ utw, const float* __restrict__ utb,
                           float* __restrict__ out) {
  int b = blockIdx.x * blockDim.x + threadIdx.x;
  if (b >= EB) return;
  const _Float16* cl = c_last + (size_t)b * 128;
  float s1 = 0.0f, s2 = 0.0f;
  for (int i = 0; i < 128; ++i) {
    float v = (float)cl[i];
    s1 += v * clw[i];
    s2 += v * ctw[i];
  }
  s1 += clb[0]; s2 += ctb[0];
  const _Float16* ul = u_last + (size_t)b * 32;
  float t1 = 0.0f, t2 = 0.0f;
  for (int i = 0; i < 32; ++i) {
    float v = (float)ul[i];
    t1 += v * tlw[i];
    t2 += v * utw[i];
  }
  t1 += tlb[0]; t2 += utb[0];
  out[0 * EB + b] = s1;
  out[1 * EB + b] = sigmoidf(s1);
  out[2 * EB + b] = s2;
  out[3 * EB + b] = t1;
  out[4 * EB + b] = sigmoidf(t1);
  out[5 * EB + b] = t2;
}

// ---------------------------------------------------------------------------
static inline void launch_gemm(const _Float16* A, int lda, const _Float16* W,
                               int ldw, const float* bias, _Float16* C, int ldc,
                               int M, int N, int K, int act, hipStream_t s) {
  int tiles = (M / 64) * (N / 32);       // waves (64x32 block per wave)
  int blocks = (tiles + 3) / 4;
  gemm_rrr_f16_blk<<<blocks, 128, 0, s>>>(A, lda, W, ldw, bias, C, ldc,
                                          M, N, K, act);
}

extern "C" void kernel_launch(void* const* d_in, const int* in_sizes, int n_in,
                              void* d_out, int out_size, void* d_ws, size_t ws_size,
                              hipStream_t stream) {
  const float* feat = (const float*)d_in[0];
  const float* E    = (const float*)d_in[2];
  const float* Qw   = (const float*)d_in[3];
  const float* Qb   = (const float*)d_in[4];
  const float* Kw   = (const float*)d_in[5];
  const float* Kb   = (const float*)d_in[6];
  const float* Vw   = (const float*)d_in[7];
  const float* Vb   = (const float*)d_in[8];
  const float* A1   = (const float*)d_in[9];
  const float* A2w  = (const float*)d_in[10];
  const float* A2b  = (const float*)d_in[11];
  const float* A3   = (const float*)d_in[12];
  const float* Tw   = (const float*)d_in[13];
  const float* Tb   = (const float*)d_in[14];
  const float* cW[6]; const float* cb[6];
  for (int i = 0; i < 6; ++i) { cW[i] = (const float*)d_in[15 + i]; cb[i] = (const float*)d_in[21 + i]; }
  const float* clw = (const float*)d_in[27]; const float* clb = (const float*)d_in[28];
  const float* ctw = (const float*)d_in[29]; const float* ctb = (const float*)d_in[30];
  const float* uW[6]; const float* ub[6];
  for (int i = 0; i < 6; ++i) { uW[i] = (const float*)d_in[31 + i]; ub[i] = (const float*)d_in[37 + i]; }
  const float* tlw = (const float*)d_in[43]; const float* tlb = (const float*)d_in[44];
  const float* utw = (const float*)d_in[45]; const float* utb = (const float*)d_in[46];
  float* out = (float*)d_out;

  // ---- workspace bump allocator ----
  size_t off = 0;
  auto alloc = [&](size_t bytes) -> char* {
    char* p = (char*)d_ws + off;
    off = (off + bytes + 255) & ~(size_t)255;
    return p;
  };
  _Float16* nx  = (_Float16*)alloc((size_t)EBF_PAD * EHU * 2);
  _Float16* Qm  = (_Float16*)alloc((size_t)EBF_PAD * EHU * 2);
  _Float16* Km  = (_Float16*)alloc((size_t)EBF_PAD * EHU * 2);
  _Float16* Vm  = (_Float16*)alloc((size_t)EBF_PAD * EHU * 2);
  _Float16* xx  = (_Float16*)alloc((size_t)EB * EF * EHU * 2);
  _Float16* cB1 = (_Float16*)alloc((size_t)EB * EHC * 2);
  _Float16* cB2 = (_Float16*)alloc((size_t)EB * EHC * 2);
  _Float16* wf  = (_Float16*)alloc((size_t)EB * 128 * 2);
  _Float16* xt  = (_Float16*)alloc((size_t)EB * 128 * 2);
  _Float16* uB1 = (_Float16*)alloc((size_t)EB * 128 * 2);
  _Float16* uB2 = (_Float16*)alloc((size_t)EB * 128 * 2);
  _Float16* Ep  = (_Float16*)alloc((size_t)128 * 128 * 2);
  _Float16* hQw = (_Float16*)alloc((size_t)EHU * EHU * 2);
  _Float16* hKw = (_Float16*)alloc((size_t)EHU * EHU * 2);
  _Float16* hVw = (_Float16*)alloc((size_t)EHU * EHU * 2);
  const int cIn[6]  = {EF * EHU, EHC, EHC, EHC, EHC / 2, EHC / 2};
  const int cOut[6] = {EHC, EHC, EHC, EHC / 2, EHC / 2, EHC / 4};
  const int uIn[6]  = {EHU, EHU, EHU, EHU, EHU / 2, EHU / 2};
  const int uOut[6] = {EHU, EHU, EHU, EHU / 2, EHU / 2, EHU / 4};
  _Float16* hcW[6]; _Float16* huW[6];
  for (int i = 0; i < 6; ++i) hcW[i] = (_Float16*)alloc((size_t)cIn[i] * cOut[i] * 2);
  for (int i = 0; i < 6; ++i) huW[i] = (_Float16*)alloc((size_t)uIn[i] * uOut[i] * 2);
  float* rnorm  = (float*)alloc((size_t)EB * EHU * 4);
  float* attT   = (float*)alloc(128 * 4);
  float* EA2    = (float*)alloc((size_t)EF * EHU * 4);
  float* logits = (float*)alloc((size_t)EBF * 4);

  // ---- weight conversion (fp32 -> fp16) ----
  auto cvt = [&](const float* s, _Float16* d, int n) {
    f32_to_f16_kern<<<(n + 255) / 256, 256, 0, stream>>>(s, d, n);
  };
  cvt(Qw, hQw, EHU * EHU);
  cvt(Kw, hKw, EHU * EHU);
  cvt(Vw, hVw, EHU * EHU);
  for (int i = 0; i < 6; ++i) cvt(cW[i], hcW[i], cIn[i] * cOut[i]);
  for (int i = 0; i < 6; ++i) cvt(uW[i], huW[i], uIn[i] * uOut[i]);
  epad_kern<<<(128 * 128 + 255) / 256, 256, 0, stream>>>(E, Ep);

  // ---- normalized representation ----
  norm_kern<<<(EB * EHU + 255) / 256, 256, 0, stream>>>(feat, E, rnorm);
  nx_kern<<<(int)(((long long)EBF * EHU + 255) / 256), 256, 0, stream>>>(feat, E, rnorm, nx);

  // ---- Q/K/V projections (WMMA, blocked) ----
  launch_gemm(nx, EHU, hQw, EHU, Qb, Qm, EHU, EBF, EHU, EHU, 0, stream);
  launch_gemm(nx, EHU, hKw, EHU, Kb, Km, EHU, EBF, EHU, EHU, 0, stream);
  launch_gemm(nx, EHU, hVw, EHU, Vb, Vm, EHU, EBF, EHU, EHU, 0, stream);

  // ---- fused attention (WMMA + LDS) ----
  attn_kern<<<EB, 224, 0, stream>>>(Qm, Km, Vm, xx);

  // ---- c-MLP (WMMA, fused bias + ELU stored for next layer) ----
  {
    const _Float16* in = xx;
    _Float16* bufs[2] = {cB1, cB2};
    int curLd = EF * EHU;
    for (int i = 0; i < 6; ++i) {
      _Float16* o = bufs[i & 1];
      launch_gemm(in, curLd, hcW[i], cIn[i], cb[i], o, cOut[i],
                  EB, cOut[i], cIn[i], 1, stream);
      in = o;
      curLd = cOut[i];
    }
  }
  const _Float16* c_last = cB2;   // layer 6 lands in cB2 (i=5 odd)

  // ---- att path -> xt ----
  att_t_kern<<<1, 128, 0, stream>>>(Tw, Tb, A1, attT);
  ea2_kern<<<(EF * EHU + 255) / 256, 256, 0, stream>>>(E, A2w, EA2);
  logits_kern<<<(EBF + 255) / 256, 256, 0, stream>>>(feat, EA2, A2b, attT, A3, logits);
  wsoft_kern<<<(EB + 255) / 256, 256, 0, stream>>>(logits, feat, wf);
  {
    int tiles = (EB / 16) * (128 / 16);
    gemm_bk_f16<<<(tiles + 3) / 4, 128, 0, stream>>>(wf, 128, Ep, 128,
                                                     xt, 128, EB, 128, 128);
  }

  // ---- u-MLP (WMMA, blocked) ----
  {
    const _Float16* in = xt;
    _Float16* bufs[2] = {uB1, uB2};
    int curLd = EHU;
    for (int i = 0; i < 6; ++i) {
      _Float16* o = bufs[i & 1];
      launch_gemm(in, curLd, huW[i], uIn[i], ub[i], o, uOut[i],
                  EB, uOut[i], uIn[i], 1, stream);
      in = o;
      curLd = uOut[i];
    }
  }
  const _Float16* u_last = uB2;

  // ---- heads ----
  heads_kern<<<(EB + 255) / 256, 256, 0, stream>>>(c_last, u_last,
                                                   clw, clb, ctw, ctb,
                                                   tlw, tlb, utw, utb, out);
}
